// Model_83141976916192
// MI455X (gfx1250) — matrix operand order repaired and checked by reading, not compiled, not run on hardware
//
#include <hip/hip_runtime.h>

// ---------------------------------------------------------------------------
// Mamba-2 SSD chunked scan, fused single pass for MI455X (gfx1250, wave32).
//
// One workgroup per (batch, head) = 128 blocks x 512 threads (16 waves).
// Each block loops over the 128 chunks of length 64, doing four 64x64x64
// matmuls per chunk with v_wmma_f32_16x16x32_bf16 and carrying the running
// inter-chunk state S (64x64) in per-wave accumulator registers, so the
// 268 MB "states" tensor never touches HBM. Total HBM traffic ~1.07 GB ->
// ~46 us roofline at 23.3 TB/s; bf16 WMMA gives ample compute headroom.
//
// v2: register-prefetch software pipeline (chunk c+1 global loads overlap
//     chunk c WMMA work) + wave-parallel cumsum scan (shfl) instead of a
//     serial thread-0 loop; exp factors computed on the fly (TRANS co-exec).
// ---------------------------------------------------------------------------

#define B_SZ   4
#define SEQ    8192
#define NH     32
#define DHEAD  64
#define DSTATE 64
#define BLK    64
#define NCHUNK (SEQ / BLK)
#define CH_STRIDE ((long)BLK * NH * DHEAD)   // floats between chunks (X/B/C)

typedef __attribute__((ext_vector_type(16))) __bf16 bf16x16;
typedef __attribute__((ext_vector_type(8)))  __bf16 bf16x8;
typedef __attribute__((ext_vector_type(8)))  float  f32x8;
typedef __attribute__((ext_vector_type(4)))  float  f32x4;

// A-operand fragment (16x32 bf16), row-major LDS [M][K], stride 64.
// Lane l: M = row; half = lane>>4.
// VGPR0-3 = K in [k0 + half*8, +8);  VGPR4-7 = K in [k0 + 16 + half*8, +8).
__device__ __forceinline__ bf16x16 frag_a(const __bf16* base, int row, int k0, int lane) {
  union { bf16x16 v; bf16x8 h[2]; } u;
  const __bf16* p = base + row * 64 + k0 + ((lane >> 4) << 3);
  u.h[0] = *(const bf16x8*)p;
  u.h[1] = *(const bf16x8*)(p + 16);
  return u.v;
}

// B-operand fragment (32x16 bf16), LDS stored [N][K], stride 64.
// Lane l: N = col; half = lane>>4; VGPR0-3 = K in [k0 + half*8, +8);  VGPR4-7 = K in [k0 + 16 + half*8, +8).
__device__ __forceinline__ bf16x16 frag_b(const __bf16* base, int col, int k0, int lane) {
  union { bf16x16 v; bf16x8 h[2]; } u;
  const __bf16* p = base + col * 64 + k0 + ((lane >> 4) << 3);
  u.h[0] = *(const bf16x8*)p;
  u.h[1] = *(const bf16x8*)(p + 16);
  return u.v;
}

// 16x16 tile of a 64x64x64 matmul: acc += A[rowTile, :] * B[:, colTile]
__device__ __forceinline__ void mm64(const __bf16* A, const __bf16* B,
                                     int row, int col, int lane, f32x8& acc) {
#pragma unroll
  for (int k0 = 0; k0 < 64; k0 += 32) {
    bf16x16 a = frag_a(A, row, k0, lane);
    bf16x16 b = frag_b(B, col, k0, lane);
    acc = __builtin_amdgcn_wmma_f32_16x16x32_bf16(false, a, false, b,
                                                  (short)0, acc, false, false);
  }
}

__global__ __launch_bounds__(512) void ssd_fused_kernel(
    const float* __restrict__ X,   // [B, SEQ, NH, DHEAD]
    const float* __restrict__ A,   // [B, SEQ, NH]
    const float* __restrict__ Bm,  // [B, SEQ, NH, DSTATE]
    const float* __restrict__ Cm,  // [B, SEQ, NH, DSTATE]
    float* __restrict__ Y)         // [B, SEQ, NH, DHEAD]
{
  const int tid  = threadIdx.x;
  const int lane = tid & 31;
  const int wave = tid >> 5;        // 0..15
  const int wm   = wave >> 2;       // output row tile (M)
  const int wn   = wave & 3;        // output col tile (N)
  const int hh   = lane >> 4;       // lane half
  const int nn   = lane & 15;       // N index within tile

  const int b = blockIdx.x / NH;
  const int h = blockIdx.x % NH;

  __shared__ __bf16 sC [64 * 64];   // C chunk,      [l][n]
  __shared__ __bf16 sB [64 * 64];   // B chunk,      [l][n]  (== [N=s][K=n] for G)
  __shared__ __bf16 sBt[64 * 64];   // decay*B^T,    [n][l]
  __shared__ __bf16 sXt[64 * 64];   // X^T chunk,    [p][l]
  __shared__ __bf16 sG [64 * 64];   // masked G,     [l][s]
  __shared__ __bf16 sS16[64 * 64];  // running state,[p][n] (bf16 copy of Sacc)
  __shared__ float  sCS[64];        // within-chunk cumsum of A

  // zero the inter-chunk state
  for (int i = tid; i < 64 * 64; i += 512) sS16[i] = (__bf16)0.0f;
  f32x8 Sacc = {};   // this wave's 16x16 tile of S[p][n] (fp32 master copy)
  __syncthreads();

  // per-thread global-load coordinates: 8 contiguous floats per row
  const int lr = tid >> 3;          // row l within chunk (0..63)
  const int q0 = (tid & 7) * 8;     // first column (0,8,..,56)

  const float* pX = X  + (((long)b * SEQ + lr) * NH + h) * (long)DHEAD  + q0;
  const float* pB = Bm + (((long)b * SEQ + lr) * NH + h) * (long)DSTATE + q0;
  const float* pC = Cm + (((long)b * SEQ + lr) * NH + h) * (long)DSTATE + q0;
  const float* pA = A  + ((long)b * SEQ + 2 * lane) * NH + h;   // wave 0 only

  // ---- prologue: prefetch chunk 0 into registers -----------------------
  union { f32x4 v[2]; float f[8]; } xu, bu, cu;
  float a0 = 0.0f, a1 = 0.0f;
  xu.v[0] = *(const f32x4*)pX;      xu.v[1] = *(const f32x4*)(pX + 4);
  bu.v[0] = *(const f32x4*)pB;      bu.v[1] = *(const f32x4*)(pB + 4);
  cu.v[0] = *(const f32x4*)pC;      cu.v[1] = *(const f32x4*)(pC + 4);
  if (wave == 0) { a0 = pA[0]; a1 = pA[NH]; }

  for (int c = 0; c < NCHUNK; ++c) {
    const long srow = (long)b * SEQ + (long)c * BLK;

    // ---- stage prefetched chunk into LDS (bf16) ------------------------
#pragma unroll
    for (int j = 0; j < 8; ++j) {
      sXt[(q0 + j) * 64 + lr] = (__bf16)xu.f[j];
      sB [lr * 64 + q0 + j]   = (__bf16)bu.f[j];
      sC [lr * 64 + q0 + j]   = (__bf16)cu.f[j];
    }

    // ---- wave-parallel inclusive cumsum of A over the chunk ------------
    if (wave == 0) {
      float s = a0 + a1;                         // pair sum, lane owns rows 2l,2l+1
#pragma unroll
      for (int d = 1; d < 32; d <<= 1) {
        float t = __shfl_up(s, d, 32);
        if (lane >= d) s += t;
      }
      const float excl = s - (a0 + a1);          // exclusive prefix of pair sums
      sCS[2 * lane]     = excl + a0;
      sCS[2 * lane + 1] = excl + a0 + a1;
    }
    __syncthreads();   // sCS + sB/sC/sXt visible

    // ---- decay-weighted B^T for the states matmul ----------------------
    {
      const float d = __expf(sCS[63] - sCS[lr]); // decay_states[l]
#pragma unroll
      for (int j = 0; j < 8; ++j) sBt[(q0 + j) * 64 + lr] = (__bf16)(bu.f[j] * d);
    }

    // ---- prefetch chunk c+1 (overlaps the whole WMMA phase) ------------
    if (c + 1 < NCHUNK) {
      pX += CH_STRIDE;  pB += CH_STRIDE;  pC += CH_STRIDE;  pA += (long)BLK * NH;
      xu.v[0] = *(const f32x4*)pX;  xu.v[1] = *(const f32x4*)(pX + 4);
      bu.v[0] = *(const f32x4*)pB;  bu.v[1] = *(const f32x4*)(pB + 4);
      cu.v[0] = *(const f32x4*)pC;  cu.v[1] = *(const f32x4*)(pC + 4);
      if (wave == 0) { a0 = pA[0]; a1 = pA[NH]; }
    }
    __syncthreads();   // all WMMA operand buffers ready

    // ---- (1) G = C * B^T, then apply segsum mask L ---------------------
    f32x8 g = {};
    mm64(sC, sB, wm * 16 + nn, wn * 16 + nn, lane, g);
#pragma unroll
    for (int v = 0; v < 8; ++v) {
      const int l = wm * 16 + v + 8 * hh;        // C/D layout: VGPR v -> M = v | v+8
      const int s = wn * 16 + nn;
      float gv = g[v];
      gv = (l >= s) ? gv * __expf(sCS[l] - sCS[s]) : 0.0f;
      sG[l * 64 + s] = (__bf16)gv;
    }
    __syncthreads();

    // ---- (2) Y_diag = (G.L) * X   (B-operand from sXt [p][s]) ----------
    f32x8 yd = {};
    mm64(sG, sXt, wm * 16 + nn, wn * 16 + nn, lane, yd);

    // ---- (3) Y_off = C * S^T      (B-operand from sS16 [p][n]) ---------
    f32x8 yo = {};
    mm64(sC, sS16, wm * 16 + nn, wn * 16 + nn, lane, yo);

    // ---- (4) states = X^T * (decay.B)  (A from sXt [p][l], B from sBt) -
    f32x8 st = {};
    mm64(sXt, sBt, wm * 16 + nn, wn * 16 + nn, lane, st);

    // ---- combine & store Y: Y[l,p] = Yd + exp(cs_l) * Yoff -------------
#pragma unroll
    for (int v = 0; v < 8; ++v) {
      const int l = wm * 16 + v + 8 * hh;
      const int p = wn * 16 + nn;
      Y[((srow + l) * NH + h) * (long)DHEAD + p] = yd[v] + __expf(sCS[l]) * yo[v];
    }

    // ---- state recurrence: S = exp(a_chunk) * S + states ---------------
    const float lam = __expf(sCS[63]);
#pragma unroll
    for (int v = 0; v < 8; ++v) Sacc[v] = lam * Sacc[v] + st[v];

    __syncthreads();   // everyone done reading sS16 (Y_off) before rewrite
#pragma unroll
    for (int v = 0; v < 8; ++v) {
      const int p = wm * 16 + v + 8 * hh;
      const int n = wn * 16 + nn;
      sS16[p * 64 + n] = (__bf16)Sacc[v];
    }
    __syncthreads();   // sS16 ready; safe to overwrite operand buffers next c
  }
}

extern "C" void kernel_launch(void* const* d_in, const int* in_sizes, int n_in,
                              void* d_out, int out_size, void* d_ws, size_t ws_size,
                              hipStream_t stream) {
  const float* X  = (const float*)d_in[0];
  const float* A  = (const float*)d_in[1];
  const float* Bm = (const float*)d_in[2];
  const float* Cm = (const float*)d_in[3];
  float* Y = (float*)d_out;
  (void)in_sizes; (void)n_in; (void)d_ws; (void)ws_size; (void)out_size;

  dim3 grid(B_SZ * NH);   // 128 workgroups, one per (batch, head)
  dim3 block(512);        // 16 waves of 32
  ssd_fused_kernel<<<grid, block, 0, stream>>>(X, A, Bm, Cm, Y);
}
